// Decoder_14516989461059
// MI455X (gfx1250) — compile-verified, matrix operand order checked by reading, not run on hardware
//
#include <hip/hip_runtime.h>

#define H      512
#define BSZ    50
#define NCHAR  131
#define CSTART 128
#define G4     2048
#define WROW   643      // N_CHAR + H
#define TSTEPS 1024
#define NBLK   16

typedef __bf16 bf16;
typedef bf16  v16bf __attribute__((ext_vector_type(16)));
typedef float v8f   __attribute__((ext_vector_type(8)));

union Frag { uint4 q[2]; v16bf v; };

// ---- workspace byte offsets ----
#define OFF_BAR   0
#define OFF_CONST 256
#define OFF_H0    8448
#define OFF_H1    (8448 + 65536)
#define OFF_W     139520            // 128 tiles * 16 kc * 32 lanes * 32B = 2 MB
#define OFF_WOUT  2236672           // 9 tiles  * 16 kc * 32 lanes * 32B = 144 KB

// ---- dynamic LDS byte offsets ----
#define L_W       0                 // 8 tiles * 16KB = 128 KB
#define L_WOUT    131072            // 16 KB
#define L_H       147456            // 64 rows * 528 bf16 (padded) = 66 KB
#define L_GATES   215040            // 64 * 128 f32 = 32 KB
#define L_C       247808            // 50*32 f32
#define L_CONST   254208            // 128 f32
#define SMEM_TOTAL 254720
#define HROW_U4   66                // padded h row stride in uint4 (528 bf16)

static __device__ __forceinline__ v8f zero8() {
    v8f z = {0.f,0.f,0.f,0.f,0.f,0.f,0.f,0.f};
    return z;
}

// gfx1250 async copy: global -> LDS, 16B per lane, tracked by ASYNCcnt.
// LDS dest: low 32 bits of the flat shared-aperture address == wave-relative LDS offset.
static __device__ __forceinline__ void async_copy16(void* lds_dst, const void* gsrc) {
    unsigned l = (unsigned)(uintptr_t)lds_dst;
    unsigned long long g = (unsigned long long)(uintptr_t)gsrc;
    asm volatile("global_load_async_to_lds_b128 %0, %1, off"
                 :: "v"(l), "v"(g) : "memory");
}
static __device__ __forceinline__ void wait_async0() {
    asm volatile("s_wait_asynccnt 0" ::: "memory");
}

// Pack W_sum = W_ih[:,NCHAR:] + W_hh into bf16 B-matrix fragment layout.
// Entry e = (T*16 + kc)*32 + lane ; lane holds 16 bf16: K = base..base+7 and base+16..base+23
__global__ void pack_w_kernel(const float* __restrict__ W_ih,
                              const float* __restrict__ W_hh,
                              bf16* __restrict__ gW) {
    int idx = blockIdx.x * blockDim.x + threadIdx.x;   // < 128*512
    int T   = idx >> 9;
    int kc  = (idx >> 5) & 15;
    int L   = idx & 31;
    int n   = L & 15, hlf = L >> 4;
    int r   = T * 16 + n;                               // gate row < 2048
    int base = kc * 32 + hlf * 8;
    bf16* dst = gW + (size_t)idx * 16;
#pragma unroll
    for (int j = 0; j < 8; ++j) {
        int k0 = base + j, k1 = base + 16 + j;
        dst[j]     = (bf16)(W_ih[r * WROW + NCHAR + k0] + W_hh[r * H + k0]);
        dst[8 + j] = (bf16)(W_ih[r * WROW + NCHAR + k1] + W_hh[r * H + k1]);
    }
}

__global__ void pack_wout_kernel(const float* __restrict__ W_out,
                                 bf16* __restrict__ gWo) {
    int idx = blockIdx.x * blockDim.x + threadIdx.x;   // < 9*512
    if (idx >= 9 * 512) return;
    int T = idx >> 9, kc = (idx >> 5) & 15, L = idx & 31;
    int n = L & 15, hlf = L >> 4;
    int r = T * 16 + n;
    int base = kc * 32 + hlf * 8;
    bf16* dst = gWo + (size_t)idx * 16;
#pragma unroll
    for (int j = 0; j < 8; ++j) {
        int k0 = base + j, k1 = base + 16 + j;
        dst[j]     = (r < NCHAR) ? (bf16)W_out[r * H + k0] : (bf16)0.f;
        dst[8 + j] = (r < NCHAR) ? (bf16)W_out[r * H + k1] : (bf16)0.f;
    }
}

// const vector, initial h (bf16, padded to 64 rows), barrier reset
__global__ void pack_misc_kernel(const float* __restrict__ W_ih,
                                 const float* __restrict__ b_ih,
                                 const float* __restrict__ b_hh,
                                 const float* __restrict__ hid,
                                 char* __restrict__ ws) {
    int idx = blockIdx.x * blockDim.x + threadIdx.x;
    bf16*  hbuf0 = (bf16*)(ws + OFF_H0);
    float* cst   = (float*)(ws + OFF_CONST);
    if (idx < 64 * 512) {
        int row = idx >> 9, col = idx & 511;
        hbuf0[idx] = (row < BSZ) ? (bf16)hid[row * H + col] : (bf16)0.f;
    } else if (idx < 64 * 512 + G4) {
        int r = idx - 64 * 512;
        cst[r] = W_ih[r * WROW + CSTART] + b_ih[r] + b_hh[r];
    } else if (idx == 64 * 512 + G4) {
        *(unsigned*)(ws + OFF_BAR) = 0u;
    }
}

extern __shared__ char smem[];

__global__ __launch_bounds__(256, 1)
void decoder_persistent(const float* __restrict__ cell,
                        const float* __restrict__ b_out,
                        float* __restrict__ out,
                        char* __restrict__ ws) {
    const int blk  = blockIdx.x;      // 0..15 : owns hidden units [32*blk, 32*blk+32)
    const int tid  = threadIdx.x;
    const int wave = tid >> 5;        // 0..7  : N-tile slot (i,i,f,f,g,g,o,o)
    const int lane = tid & 31;
    const int nlo  = lane & 15;
    const int hlf  = lane >> 4;

    bf16*  lds_W     = (bf16*)(smem + L_W);
    bf16*  lds_h     = (bf16*)(smem + L_H);
    float* lds_gates = (float*)(smem + L_GATES);
    float* lds_c     = (float*)(smem + L_C);
    float* lds_const = (float*)(smem + L_CONST);

    unsigned*    bar   = (unsigned*)(ws + OFF_BAR);
    const float* cst   = (const float*)(ws + OFF_CONST);
    bf16*        hbuf0 = (bf16*)(ws + OFF_H0);
    bf16*        hbuf1 = (bf16*)(ws + OFF_H1);
    const uint4* gW    = (const uint4*)(ws + OFF_W);
    const uint4* gWo   = (const uint4*)(ws + OFF_WOUT);

    // ---- one-time: park weight slices in LDS via async DMA (resident for all 1024 steps) ----
    for (int s = 0; s < 8; ++s) {
        int gT = ((s >> 1) * 32) + 2 * blk + (s & 1);   // global gate N-tile
        const uint4* src = gW + (size_t)gT * 1024;      // 16 KB per tile
        uint4* dst = ((uint4*)lds_W) + s * 1024;
        for (int i = tid; i < 1024; i += 256) async_copy16(dst + i, src + i);
    }
    if (blk < 9) {
        const uint4* src = gWo + (size_t)blk * 1024;
        uint4* dst = (uint4*)(smem + L_WOUT);
        for (int i = tid; i < 1024; i += 256) async_copy16(dst + i, src + i);
    }
    {   // initial h -> LDS (padded rows)
        const uint4* src = (const uint4*)hbuf0;
        uint4* dst = (uint4*)lds_h;
        for (int i = tid; i < 4096; i += 256) {
            int row = i >> 6, c = i & 63;
            async_copy16(dst + row * HROW_U4 + c, src + i);
        }
    }
    for (int i = tid; i < 128; i += 256) {
        int s = i >> 4, n = i & 15;
        int gT = ((s >> 1) * 32) + 2 * blk + (s & 1);
        lds_const[i] = cst[gT * 16 + n];
    }
    for (int i = tid; i < BSZ * 32; i += 256) {
        int b = i >> 5, u = i & 31;
        lds_c[i] = cell[b * H + blk * 32 + u];
    }
    wait_async0();
    __syncthreads();

    const uint4* hq = (const uint4*)lds_h;
    const uint4* wq = (const uint4*)lds_W;
    const uint4* woq = (const uint4*)(smem + L_WOUT);

    for (int t = 0; t < TSTEPS; ++t) {
        // ================= gates GEMM: 4 M-tiles x this wave's N-tile, K=512 =================
        v8f acc[4];
#pragma unroll
        for (int m = 0; m < 4; ++m) acc[m] = zero8();

        for (int kc = 0; kc < 16; ++kc) {
            Frag bf;
            const uint4* bp = wq + ((size_t)(wave * 16 + kc) * 32 + lane) * 2;
            bf.q[0] = bp[0]; bf.q[1] = bp[1];
            int kb = kc * 32 + hlf * 8;                 // element offset within h row
            Frag a[4];
#pragma unroll
            for (int m = 0; m < 4; ++m) {
                int base = (m * 16 + nlo) * HROW_U4 + (kb >> 3);
                a[m].q[0] = hq[base];
                a[m].q[1] = hq[base + 2];
            }
#pragma unroll
            for (int m = 0; m < 4; ++m)
                acc[m] = __builtin_amdgcn_wmma_f32_16x16x32_bf16(
                    false, a[m].v, false, bf.v, (short)0, acc[m], false, false);
        }

        // ---- stage C tiles (+const) into LDS gates buffer ----
        {
            int col = wave * 16 + nlo;
            float cadd = lds_const[col];
#pragma unroll
            for (int m = 0; m < 4; ++m)
#pragma unroll
                for (int v = 0; v < 8; ++v)
                    lds_gates[(m * 16 + v + 8 * hlf) * 128 + col] = acc[m][v] + cadd;
        }
        __syncthreads();

        // ================= pointwise LSTM cell, write h_new (bf16) to global =================
        bf16* wdst = (t & 1) ? hbuf0 : hbuf1;
        for (int i = tid; i < BSZ * 32; i += 256) {
            int b = i >> 5, u = i & 31;
            float gi = lds_gates[b * 128 + u];
            float gf = lds_gates[b * 128 + 32 + u];
            float gg = lds_gates[b * 128 + 64 + u];
            float go = lds_gates[b * 128 + 96 + u];
            float si = 1.f / (1.f + __expf(-gi));
            float sf = 1.f / (1.f + __expf(-gf));
            float tg = tanhf(gg);
            float so = 1.f / (1.f + __expf(-go));
            float c  = sf * lds_c[i] + si * tg;
            lds_c[i] = c;
            float hn = so * tanhf(c);
            wdst[b * H + blk * 32 + u] = (bf16)hn;
        }

        // ================= device-wide barrier (16 co-resident blocks) =================
        __threadfence();
        __syncthreads();
        if (tid == 0) {
            __hip_atomic_fetch_add(bar, 1u, __ATOMIC_ACQ_REL, __HIP_MEMORY_SCOPE_AGENT);
            unsigned target = (unsigned)(NBLK * (t + 1));
            while (__hip_atomic_load(bar, __ATOMIC_ACQUIRE, __HIP_MEMORY_SCOPE_AGENT) < target)
                __builtin_amdgcn_s_sleep(2);
        }
        __syncthreads();
        __threadfence();

        // ---- refresh LDS h with h_new via async DMA (used by output GEMM + next step) ----
        {
            const uint4* src = (const uint4*)wdst;
            uint4* dst = (uint4*)lds_h;
            for (int i = tid; i < 4096; i += 256) {
                int row = i >> 6, c = i & 63;
                async_copy16(dst + row * HROW_U4 + c, src + i);
            }
            wait_async0();
        }
        __syncthreads();

        // ================= output GEMM: blocks 0..8 each own one 16-col tile of out =================
        if (blk < 9 && wave < 4) {
            int mt = wave;
            v8f oacc = zero8();
            for (int kc = 0; kc < 16; ++kc) {
                Frag bf;
                const uint4* bp = woq + ((size_t)(kc * 32 + lane)) * 2;
                bf.q[0] = bp[0]; bf.q[1] = bp[1];
                int kb = kc * 32 + hlf * 8;
                Frag a;
                int base = (mt * 16 + nlo) * HROW_U4 + (kb >> 3);
                a.q[0] = hq[base];
                a.q[1] = hq[base + 2];
                oacc = __builtin_amdgcn_wmma_f32_16x16x32_bf16(
                    false, a.v, false, bf.v, (short)0, oacc, false, false);
            }
            int col = blk * 16 + nlo;
            if (col < NCHAR) {
                float bo = b_out[col];
#pragma unroll
                for (int v = 0; v < 8; ++v) {
                    int row = mt * 16 + v + 8 * hlf;
                    if (row < BSZ)
                        out[((size_t)row * NCHAR + col) * TSTEPS + t] = oacc[v] + bo;
                }
            }
        }
    }
}

extern "C" void kernel_launch(void* const* d_in, const int* in_sizes, int n_in,
                              void* d_out, int out_size, void* d_ws, size_t ws_size,
                              hipStream_t stream) {
    (void)in_sizes; (void)n_in; (void)out_size; (void)ws_size;
    const float* hid   = (const float*)d_in[0];
    const float* cellp = (const float*)d_in[1];
    const float* W_ih  = (const float*)d_in[2];
    const float* W_hh  = (const float*)d_in[3];
    const float* b_ih  = (const float*)d_in[4];
    const float* b_hh  = (const float*)d_in[5];
    const float* W_out = (const float*)d_in[6];
    const float* b_out = (const float*)d_in[7];
    float* out = (float*)d_out;
    char*  ws  = (char*)d_ws;

    bf16* gW  = (bf16*)(ws + OFF_W);
    bf16* gWo = (bf16*)(ws + OFF_WOUT);

    pack_w_kernel<<<256, 256, 0, stream>>>(W_ih, W_hh, gW);
    pack_wout_kernel<<<18, 256, 0, stream>>>(W_out, gWo);
    pack_misc_kernel<<<137, 256, 0, stream>>>(W_ih, b_ih, b_hh, hid, ws);

    hipFuncSetAttribute((const void*)decoder_persistent,
                        hipFuncAttributeMaxDynamicSharedMemorySize, SMEM_TOTAL);
    decoder_persistent<<<NBLK, 256, SMEM_TOTAL, stream>>>(cellp, b_out, out, ws);
}
